// AdditiveUnpoolingWrapper_12627203851175
// MI455X (gfx1250) — matrix operand order repaired and checked by reading, not compiled, not run on hardware
//
#include <hip/hip_runtime.h>
#include <hip/hip_bf16.h>
#include <stdint.h>

// MI455X / gfx1250, wave32. D = A*B + C via V_WMMA_F32_16X16X32_BF16.
// fp32 inputs split into bf16 hi/lo; 3-term WMMA recovers ~fp32 accuracy.
// Memory-bound problem (~1 GB HBM traffic -> ~43 us at 23.3 TB/s); bf16 WMMA
// keeps the 3x-FLOP GEMMs (~103 GFLOP) well under the memory floor.

typedef __attribute__((ext_vector_type(16))) __bf16 v16bf;
typedef __attribute__((ext_vector_type(8)))  float  v8f;
typedef __attribute__((ext_vector_type(4)))  float  v4f;

#define LN_EPS 1e-5f

__device__ __forceinline__ float gelu_tanh(float y) {
    float u = 0.7978845608028654f * (y + 0.044715f * y * y * y);
    return 0.5f * y * (1.0f + tanhf(u));
}

__device__ __forceinline__ void split_bf16(float x, __bf16& h, __bf16& l) {
    h = (__bf16)x;                 // RNE to bf16
    l = (__bf16)(x - (float)h);    // residual
}

__device__ __forceinline__ uint32_t pack2(__bf16 a, __bf16 b) {
    uint16_t ua = __builtin_bit_cast(uint16_t, a);
    uint16_t ub = __builtin_bit_cast(uint16_t, b);
    return ((uint32_t)ub << 16) | (uint32_t)ua;
}

// One wave computes a 16-row x 128-col tile of out = GELU(LN(X @ W + bias)) [+ gather add].
// Block = 256 threads = 8 waves = 128 rows sharing one LDS-staged weight copy
// (stage-1 LDS is 128 KB -> 2 WGs per 320 KB WGP -> 16 waves/WGP).
template <int KTILES, bool GATHER>
__global__ void __launch_bounds__(256)
fused_proj_kernel(const float* __restrict__ X,      // [rows, KTILES*32]
                  const float* __restrict__ W,      // [KTILES*32, 128] row-major
                  const float* __restrict__ bias,   // [128]
                  const float* __restrict__ ln_g,   // [128]
                  const float* __restrict__ ln_b,   // [128]
                  const int*   __restrict__ sub_idx,// [rows] (GATHER only)
                  const float* __restrict__ pooled, // [M,128] (GATHER only)
                  float* __restrict__ out)          // [rows,128]
{
    // Weights staged in LDS, pre-swizzled into WMMA B-fragment layout:
// frag[kt][nt][lane] = v16bf: lane&15 = column, kb = 8*(lane>>4),
// dword r<4 = {K=kb+2r, kb+2r+1}, dword r>=4 = {K=16+kb+2(r-4), +1}.
    __shared__ v16bf sBhi[KTILES][8][32];
    __shared__ v16bf sBlo[KTILES][8][32];

    const int tid = threadIdx.x;
    const int K   = KTILES * 32;

    for (int e = tid; e < KTILES * 8 * 32 * 8; e += 256) {
        int reg = e & 7;
        int ln2 = (e >> 3) & 31;
        int nt  = (e >> 8) & 7;
        int kt  = e >> 11;
        int n   = nt * 16 + (ln2 & 15);
        int kk  = kt * 32 + ((ln2 >> 4) << 3) + ((reg >> 2) << 4) + 2 * (reg & 3);
        float w0 = W[(long)kk * 128 + n];
        float w1 = W[(long)(kk + 1) * 128 + n];
        __bf16 h0, l0, h1, l1;
        split_bf16(w0, h0, l0);
        split_bf16(w1, h1, l1);
        ((uint32_t*)&sBhi[kt][nt][ln2])[reg] = pack2(h0, h1);
        ((uint32_t*)&sBlo[kt][nt][ln2])[reg] = pack2(l0, l1);
    }
    __syncthreads();

    const int lane = tid & 31;
    const int wv   = tid >> 5;
    const long row0 = (long)blockIdx.x * 128 + (long)wv * 16;
    const int mrow = lane & 15;                 // A-matrix row within tile
    const int kb   = (lane < 16) ? 0 : 8;       // A-layout K base for this lane half
    const float* arow = X + (row0 + mrow) * (long)K + kb;

    v8f c[8];
    #pragma unroll
    for (int nt = 0; nt < 8; ++nt)
        #pragma unroll
        for (int r = 0; r < 8; ++r) c[nt][r] = 0.0f;

    // Double-buffered A-tile loads: issue kt+1's global b128 loads before the
    // kt conversions/WMMAs so the load latency overlaps the matrix work.
    v4f cur0, cur1, cur2, cur3;
    {
        const float* ap = arow;
        cur0 = *(const v4f*)(ap);
        cur1 = *(const v4f*)(ap + 4);
        cur2 = *(const v4f*)(ap + 16);
        cur3 = *(const v4f*)(ap + 20);
    }

    for (int kt = 0; kt < KTILES; ++kt) {
        v4f nxt0, nxt1, nxt2, nxt3;
        if (kt + 1 < KTILES) {
            const float* ap = arow + (kt + 1) * 32;
            nxt0 = *(const v4f*)(ap);
            nxt1 = *(const v4f*)(ap + 4);
            nxt2 = *(const v4f*)(ap + 16);
            nxt3 = *(const v4f*)(ap + 20);
        }

        // A fragment (16-bit A layout): regs 0-3 = K kb..kb+7, regs 4-7 = K kb+16..kb+23
        float fl[16];
        #pragma unroll
        for (int i = 0; i < 4; ++i) {
            fl[i] = cur0[i]; fl[4 + i] = cur1[i]; fl[8 + i] = cur2[i]; fl[12 + i] = cur3[i];
        }
        v16bf ahi, alo;
        #pragma unroll
        for (int i = 0; i < 16; ++i) {
            __bf16 h, l;
            split_bf16(fl[i], h, l);
            ahi[i] = h; alo[i] = l;
        }

        // Process 4 column tiles per group: load all 8 fragments into distinct
        // registers, then interleave the 12 WMMAs across 4 accumulators.
        // Avoids the BF16 WMMA->VALU WAR hazard (4 co-exec NOPs) seen when a
        // fragment register is reloaded right after the WMMA that read it, and
        // gives same-accumulator WMMAs a 3-instruction spacing.
        #pragma unroll
        for (int ntg = 0; ntg < 8; ntg += 4) {
            v16bf bhi0 = sBhi[kt][ntg + 0][lane];
            v16bf bhi1 = sBhi[kt][ntg + 1][lane];
            v16bf bhi2 = sBhi[kt][ntg + 2][lane];
            v16bf bhi3 = sBhi[kt][ntg + 3][lane];
            v16bf blo0 = sBlo[kt][ntg + 0][lane];
            v16bf blo1 = sBlo[kt][ntg + 1][lane];
            v16bf blo2 = sBlo[kt][ntg + 2][lane];
            v16bf blo3 = sBlo[kt][ntg + 3][lane];

            c[ntg + 0] = __builtin_amdgcn_wmma_f32_16x16x32_bf16(false, ahi, false, bhi0, (short)0, c[ntg + 0], false, false);
            c[ntg + 1] = __builtin_amdgcn_wmma_f32_16x16x32_bf16(false, ahi, false, bhi1, (short)0, c[ntg + 1], false, false);
            c[ntg + 2] = __builtin_amdgcn_wmma_f32_16x16x32_bf16(false, ahi, false, bhi2, (short)0, c[ntg + 2], false, false);
            c[ntg + 3] = __builtin_amdgcn_wmma_f32_16x16x32_bf16(false, ahi, false, bhi3, (short)0, c[ntg + 3], false, false);

            c[ntg + 0] = __builtin_amdgcn_wmma_f32_16x16x32_bf16(false, ahi, false, blo0, (short)0, c[ntg + 0], false, false);
            c[ntg + 1] = __builtin_amdgcn_wmma_f32_16x16x32_bf16(false, ahi, false, blo1, (short)0, c[ntg + 1], false, false);
            c[ntg + 2] = __builtin_amdgcn_wmma_f32_16x16x32_bf16(false, ahi, false, blo2, (short)0, c[ntg + 2], false, false);
            c[ntg + 3] = __builtin_amdgcn_wmma_f32_16x16x32_bf16(false, ahi, false, blo3, (short)0, c[ntg + 3], false, false);

            c[ntg + 0] = __builtin_amdgcn_wmma_f32_16x16x32_bf16(false, alo, false, bhi0, (short)0, c[ntg + 0], false, false);
            c[ntg + 1] = __builtin_amdgcn_wmma_f32_16x16x32_bf16(false, alo, false, bhi1, (short)0, c[ntg + 1], false, false);
            c[ntg + 2] = __builtin_amdgcn_wmma_f32_16x16x32_bf16(false, alo, false, bhi2, (short)0, c[ntg + 2], false, false);
            c[ntg + 3] = __builtin_amdgcn_wmma_f32_16x16x32_bf16(false, alo, false, bhi3, (short)0, c[ntg + 3], false, false);
        }

        cur0 = nxt0; cur1 = nxt1; cur2 = nxt2; cur3 = nxt3;
    }

    // Epilogue in registers. C/D layout: lane&15 = column, reg r -> row
    // m = r (lanes 0-15) or r+8 (lanes 16-31).
    const int ncol0 = lane & 15;
    float gcol[8], bcol[8];
    #pragma unroll
    for (int nt = 0; nt < 8; ++nt) {
        int n = nt * 16 + ncol0;
        float bs = bias[n];
        gcol[nt] = ln_g[n];
        bcol[nt] = ln_b[n];
        #pragma unroll
        for (int r = 0; r < 8; ++r) c[nt][r] += bs;
    }

    float mean[8], rstd[8];
    #pragma unroll
    for (int r = 0; r < 8; ++r) {
        float s = 0.0f, q = 0.0f;
        #pragma unroll
        for (int nt = 0; nt < 8; ++nt) { float v = c[nt][r]; s += v; q += v * v; }
        // reduce across the 16-lane half-group (xor of bits 0..3 stays in-group)
        #pragma unroll
        for (int d = 1; d < 16; d <<= 1) {
            s += __shfl_xor(s, d, 32);
            q += __shfl_xor(q, d, 32);
        }
        float mu  = s * 0.0078125f;
        float var = q * 0.0078125f - mu * mu;
        mean[r] = mu;
        rstd[r] = rsqrtf(var + LN_EPS);
    }

    #pragma unroll
    for (int r = 0; r < 8; ++r) {
        long m = row0 + r + ((lane < 16) ? 0 : 8);
        long gbase = 0;
        if constexpr (GATHER) gbase = (long)sub_idx[m] * 128;
        #pragma unroll
        for (int nt = 0; nt < 8; ++nt) {
            int n = nt * 16 + ncol0;
            float x = (c[nt][r] - mean[r]) * rstd[r];
            float y = x * gcol[nt] + bcol[nt];
            float g = gelu_tanh(y);
            if constexpr (GATHER) g += pooled[gbase + n];
            out[m * 128 + n] = g;
        }
    }
}

extern "C" void kernel_launch(void* const* d_in, const int* in_sizes, int n_in,
                              void* d_out, int out_size, void* d_ws, size_t ws_size,
                              hipStream_t stream) {
    const float* residual = (const float*)d_in[0];   // [N,128]
    const float* down     = (const float*)d_in[1];   // [M,256]
    const float* W_down   = (const float*)d_in[2];   // [256,128]
    const float* b_down   = (const float*)d_in[3];
    const float* g_down   = (const float*)d_in[4];
    const float* lb_down  = (const float*)d_in[5];
    const float* W_skip   = (const float*)d_in[6];   // [128,128]
    const float* b_skip   = (const float*)d_in[7];
    const float* g_skip   = (const float*)d_in[8];
    const float* lb_skip  = (const float*)d_in[9];
    const int*   sub_idx  = (const int*)d_in[10];    // [N]
    float*       out      = (float*)d_out;           // [N,128]
    float*       proj_dn  = (float*)d_ws;            // [M,128] staging (128 MB)

    const int M = in_sizes[1] / 256;  // 262144
    const int N = in_sizes[0] / 128;  // 524288

    // Stage 1: proj_down = GELU(LN(down @ W_down + b)) -> workspace
    fused_proj_kernel<8, false><<<M / 128, 256, 0, stream>>>(
        down, W_down, b_down, g_down, lb_down, nullptr, nullptr, proj_dn);

    // Stage 2: out = GELU(LN(residual @ W_skip + b)) + proj_down[subbuck_idx]
    fused_proj_kernel<4, true><<<N / 128, 256, 0, stream>>>(
        residual, W_skip, b_skip, g_skip, lb_skip, sub_idx, proj_dn, out);
}
